// XConv_9088150798612
// MI455X (gfx1250) — matrix operand order repaired and checked by reading, not compiled, not run on hardware
//
#include <hip/hip_runtime.h>
#include <hip/hip_bf16.h>
#include <math.h>

// ---------------- constants ----------------
#define BQ     4
#define NCQ    2048
#define NPQ    8192
#define KNN    16
#define CDEL   64
#define CFQ    128
#define CIN    192          // CDEL + CF
#define CINK   3072         // KNN * CIN  (flattened fa row)
#define COUT   256
#define DMQ    4
#define NPTS   (BQ*NCQ)     // 8192 points total
#define KT1    96           // 3072 / 32 k-steps for wx0 gemm
#define KT2    24           // 768 / 32 k-steps for pw gemm
#define CDW    768          // CIN * DM
#define STAGE_H 16384       // halves per LDS stage buffer (2 k-slices of 8192)

typedef _Float16 v8h  __attribute__((ext_vector_type(8)));
typedef _Float16 v16h __attribute__((ext_vector_type(16)));
typedef float    v8f  __attribute__((ext_vector_type(8)));

__device__ __forceinline__ float elu_f(float x) {
    return x > 0.f ? x : (__expf(x) - 1.f);
}

// ---------------- 1) kNN: one wave per center, top-32 sorted, keep ranks 0,2,..,30 ----
__global__ void knn_kernel(const float* __restrict__ center,
                           const float* __restrict__ coords,
                           int* __restrict__ idx_out) {
    const int blk  = blockIdx.x;            // 0 .. NPTS-1
    const int b    = blk >> 11;
    const int n    = blk & (NCQ - 1);
    const int lane = threadIdx.x;           // wave32

    __shared__ float s_d[32 * 32];
    __shared__ int   s_i[32 * 32];
    __shared__ int   s_out[KNN];

    const float cx = center[(size_t)b * 3 * NCQ + 0 * NCQ + n];
    const float cy = center[(size_t)b * 3 * NCQ + 1 * NCQ + n];
    const float cz = center[(size_t)b * 3 * NCQ + 2 * NCQ + n];
    const float* px = coords + (size_t)b * 3 * NPQ;
    const float* py = px + NPQ;
    const float* pz = py + NPQ;

    float* md = s_d + lane * 32;
    int*   mi = s_i + lane * 32;
    for (int j = 0; j < 32; ++j) { md[j] = 3.4e38f; mi[j] = 0; }

    // each lane scans 256 points (stride-32, coalesced)
    for (int it = 0; it < NPQ / 32; ++it) {
        const int m = lane + (it << 5);
        const float dx = px[m] - cx, dy = py[m] - cy, dz = pz[m] - cz;
        const float d2 = dx * dx + dy * dy + dz * dz;
        if (d2 < md[31]) {
            int pos = 31;
            while (pos > 0 && md[pos - 1] > d2) {
                md[pos] = md[pos - 1]; mi[pos] = mi[pos - 1]; --pos;
            }
            md[pos] = d2; mi[pos] = m;
        }
    }

    // 32-round wave-min merge of the 32 sorted lists -> globally sorted top-32
    int cur = 0;
    for (int r = 0; r < 32; ++r) {
        float bd = (cur < 32) ? md[cur] : 3.4e38f;
        int   bi = (cur < 32) ? mi[cur] : 0;
        int   bl = lane;
        for (int off = 16; off > 0; off >>= 1) {
            const float od = __shfl_xor(bd, off, 32);
            const int   oi = __shfl_xor(bi, off, 32);
            const int   ol = __shfl_xor(bl, off, 32);
            if (od < bd || (od == bd && ol < bl)) { bd = od; bi = oi; bl = ol; }
        }
        if (bl == lane) ++cur;                       // winner pops its head
        if ((r & 1) == 0 && lane == 0) s_out[r >> 1] = bi;  // dilation ::2
    }
    if (lane < KNN) idx_out[(size_t)blk * KNN + lane] = s_out[lane];
}

// ---------------- 2) build fa rows (f16, WMMA-A friendly k-major layout) --------------
__global__ void build_fa_kernel(const float* __restrict__ coords,
                                const float* __restrict__ center,
                                const float* __restrict__ features,
                                const float* __restrict__ w_d1, const float* __restrict__ b_d1,
                                const float* __restrict__ w_d2, const float* __restrict__ b_d2,
                                const int* __restrict__ idx,
                                _Float16* __restrict__ fa) {
    const int p = blockIdx.x;
    const int b = p >> 11;
    const int n = p & (NCQ - 1);
    const int t = threadIdx.x;     // 128 threads

    __shared__ int   s_idx[KNN];
    __shared__ float s_loc[KNN][3];
    __shared__ float s_h[KNN][CDEL];

    if (t < KNN) s_idx[t] = idx[(size_t)p * KNN + t];
    __syncthreads();

    if (t < 48) {
        const int k = t / 3, comp = t % 3;
        const float pc = coords[(size_t)b * 3 * NPQ + (size_t)comp * NPQ + s_idx[k]];
        const float cc = center[(size_t)b * 3 * NCQ + (size_t)comp * NCQ + n];
        s_loc[k][comp] = pc - cc;
    }
    __syncthreads();

    {   // h1 = elu(local @ w_d1 + b_d1)   (16 x 64)
        const int k = t >> 3, d0 = (t & 7) * 8;
        for (int j = 0; j < 8; ++j) {
            const int d = d0 + j;
            float s = b_d1[d];
            for (int c = 0; c < 3; ++c) s += s_loc[k][c] * w_d1[c * CDEL + d];
            s_h[k][d] = elu_f(s);
        }
    }
    __syncthreads();

    {   // h2 = elu(h1 @ w_d2 + b_d2) -> fa[:, 0:64]
        const int k = t >> 3, d0 = (t & 7) * 8;
        for (int j = 0; j < 8; ++j) {
            const int d = d0 + j;
            float s = b_d2[d];
            for (int c = 0; c < CDEL; ++c) s += s_h[k][c] * w_d2[c * CDEL + d];
            fa[(size_t)p * CINK + (size_t)k * CIN + d] = (_Float16)elu_f(s);
        }
    }

    {   // gathered features -> fa[:, 64:192]
        const int c = t;   // 0..127
        for (int k = 0; k < KNN; ++k) {
            const float v = features[(size_t)b * CFQ * NPQ + (size_t)c * NPQ + s_idx[k]];
            fa[(size_t)p * CINK + (size_t)k * CIN + CDEL + c] = (_Float16)v;
        }
    }
}

// ---------------- 3) weight packing into B-fragment order ----------------------------
// packed[((kt*NT + nt)*32 + lane)*16 + h] = W[N = nt*16 + (lane&15), K = kt*32 + (lane>=16?16:0) + h]
__global__ void pack_w0_kernel(const float* __restrict__ wx0, _Float16* __restrict__ w0p) {
    const int t = blockIdx.x * blockDim.x + threadIdx.x;   // KT1*16*32*16 = 786432
    if (t >= KT1 * 16 * 32 * 16) return;
    const int h    = t & 15;
    const int lane = (t >> 4) & 31;
    const int nt   = (t >> 9) & 15;
    const int kt   = t >> 13;
    const int N  = nt * 16 + (lane & 15);
    const int Kk = kt * 32 + ((lane >= 16) ? 16 : 0) + h;
    w0p[t] = (_Float16)wx0[(size_t)N * CINK + Kk];         // wx0 is (256, 3072) row-major
}

__global__ void pack_wpw_kernel(const float* __restrict__ w_pw, _Float16* __restrict__ wp) {
    const int t = blockIdx.x * blockDim.x + threadIdx.x;   // KT2*16*32*16 = 196608
    if (t >= KT2 * 16 * 32 * 16) return;
    const int h    = t & 15;
    const int lane = (t >> 4) & 31;
    const int nt   = (t >> 9) & 15;
    const int kt   = t >> 13;
    const int N  = nt * 16 + (lane & 15);
    const int Kk = kt * 32 + ((lane >= 16) ? 16 : 0) + h;
    wp[t] = (_Float16)w_pw[(size_t)Kk * COUT + N];         // w_pw is (768, 256) row-major
}

// ---------------- 4+6) staged async-LDS WMMA GEMM (shared main loop) ------------------
// C(8192 x 256) = A(8192 x KD) x Bp, B staged through LDS with CDNA5 async copies.
// Stage = 2 k-slices (32KB). Symmetric loop body (boff XOR toggle, no unroll) keeps
// accumulators in fixed registers (no v_mov shuffles / spills).
template<int KD, int NSTAGE, bool FINAL>
__global__ void __launch_bounds__(256, 1)
gemm_async_kernel(const _Float16* __restrict__ A,
                  const _Float16* __restrict__ Bp,
                  const float* __restrict__ bias,
                  float* __restrict__ outp) {
    __shared__ _Float16 s_b[2 * STAGE_H];   // 2 x 32KB stage buffers

    const int tid   = threadIdx.x;
    const int lane  = tid & 31;
    const int wave  = tid >> 5;             // 8 waves
    const int mtile = wave >> 1;            // 4 M-tiles of 16 rows
    const int nh    = wave & 1;             // N half (128 cols each)
    const int p0    = blockIdx.x * 64 + mtile * 16;
    const int lr    = lane & 15;
    const bool hi   = lane >= 16;

    const _Float16* arow = A + (size_t)(p0 + lr) * KD + (hi ? 16 : 0);

    // cooperative async copy of one 32KB B stage into LDS at half-offset `boff`
    auto copy_stage = [&](unsigned boff, int st) {
        const _Float16* src = Bp + (size_t)st * STAGE_H;
        #pragma unroll
        for (int j = 0; j < 8; ++j) {
            const int chunk = tid + j * 256;                  // 2048 chunks of 16B
            unsigned loff = (unsigned)(uintptr_t)(&s_b[boff + chunk * 8]);
            unsigned long long gaddr = (unsigned long long)(uintptr_t)(src + chunk * 8);
            asm volatile("global_load_async_to_lds_b128 %0, %1, off"
                         :: "v"(loff), "v"(gaddr) : "memory");
        }
    };

    copy_stage(0u, 0);
    asm volatile("s_wait_asynccnt 0x0" ::: "memory");
    __syncthreads();

    v8f acc[8] = {};
    unsigned boff = 0;
    #pragma clang loop unroll(disable)
    for (int st = 0; st < NSTAGE; ++st) {
        if (st + 1 < NSTAGE) copy_stage(boff ^ STAGE_H, st + 1);  // async prefetch

        #pragma unroll
        for (int kk = 0; kk < 2; ++kk) {
            const int kt = st * 2 + kk;
            __builtin_prefetch(arow + (kt + 4) * 32, 0, 3);
            const v8h c0 = *(const v8h*)(arow + kt * 32);
            const v8h c1 = *(const v8h*)(arow + kt * 32 + 8);
            v16h a;
            #pragma unroll
            for (int i = 0; i < 8; ++i) { a[i] = c0[i]; a[i + 8] = c1[i]; }

            const _Float16* bbase = s_b + boff + kk * 8192 + ((nh * 8) * 32 + lane) * 16;
            #pragma unroll
            for (int nt = 0; nt < 8; ++nt) {
                const v16h bf = *(const v16h*)(bbase + nt * 512);
                acc[nt] = __builtin_amdgcn_wmma_f32_16x16x32_f16(
                    false, a, false, bf, (short)0, acc[nt], false, false);
            }
        }

        asm volatile("s_wait_asynccnt 0x0" ::: "memory");
        __syncthreads();      // next stage landed; all waves done reading current
        boff ^= STAGE_H;
    }

    const int rowbase = p0 + (hi ? 8 : 0);
    #pragma unroll
    for (int nt = 0; nt < 8; ++nt) {
        const int col = (nh * 8 + nt) * 16 + lr;
        if constexpr (FINAL) {
            const float bv = bias[col];
            #pragma unroll
            for (int r = 0; r < 8; ++r) {
                const int prow = rowbase + r;
                const int b = prow >> 11;
                const int n = prow & (NCQ - 1);
                outp[(size_t)b * COUT * NCQ + (size_t)col * NCQ + n] = elu_f(acc[nt][r] + bv);
            }
        } else {
            #pragma unroll
            for (int r = 0; r < 8; ++r)
                outp[(size_t)(rowbase + r) * COUT + col] = elu_f(acc[nt][r]);
        }
    }
}

// ---------------- 5) middle: wx1/wx2 chains, fx = trans@fa, dw ------------------------
__global__ void middle_kernel(const float* __restrict__ t_in,
                              const _Float16* __restrict__ fa,
                              const float* __restrict__ wx1,
                              const float* __restrict__ wx2,
                              const float* __restrict__ w_dw,
                              _Float16* __restrict__ dwv) {
    const int p = blockIdx.x;
    const int t = threadIdx.x;   // 256 threads
    __shared__ float s_t[256], s_t2[256], s_t3[256];

    s_t[t] = t_in[(size_t)p * 256 + t];
    __syncthreads();

    {   // T2[c,m] = elu(sum_k T1[k,c] * wx1[c,m,k]); stored so next stage reads [k',c']
        const int c = t >> 4, m = t & 15;
        float s = 0.f;
        for (int k = 0; k < 16; ++k) s += s_t[k * 16 + c] * wx1[(c * 16 + m) * 16 + k];
        s_t2[t] = elu_f(s);
    }
    __syncthreads();
    {   // trans[c',m'] = sum_k' T2[k',c'] * wx2[c',m',k']
        const int c = t >> 4, m = t & 15;
        float s = 0.f;
        for (int k = 0; k < 16; ++k) s += s_t2[k * 16 + c] * wx2[(c * 16 + m) * 16 + k];
        s_t3[t] = s;
    }
    __syncthreads();

    if (t < CIN) {   // fx column c, then dw
        const int c = t;
        float fj[16], fx[16];
        #pragma unroll
        for (int j = 0; j < 16; ++j) fj[j] = (float)fa[(size_t)p * CINK + (size_t)j * CIN + c];
        #pragma unroll
        for (int i = 0; i < 16; ++i) {
            float s = 0.f;
            #pragma unroll
            for (int j = 0; j < 16; ++j) s += s_t3[i * 16 + j] * fj[j];
            fx[i] = s;
        }
        #pragma unroll
        for (int m = 0; m < DMQ; ++m) {
            float s = 0.f;
            #pragma unroll
            for (int k = 0; k < 16; ++k) s += fx[k] * w_dw[c * 64 + m * 16 + k];
            dwv[(size_t)p * CDW + c * DMQ + m] = (_Float16)s;
        }
    }
}

// ---------------- launch --------------------------------------------------------------
extern "C" void kernel_launch(void* const* d_in, const int* in_sizes, int n_in,
                              void* d_out, int out_size, void* d_ws, size_t ws_size,
                              hipStream_t stream) {
    const float* center   = (const float*)d_in[0];
    const float* coords   = (const float*)d_in[1];
    const float* features = (const float*)d_in[2];
    const float* w_d1     = (const float*)d_in[3];
    const float* b_d1     = (const float*)d_in[4];
    const float* w_d2     = (const float*)d_in[5];
    const float* b_d2     = (const float*)d_in[6];
    const float* wx0      = (const float*)d_in[7];
    const float* wx1      = (const float*)d_in[8];
    const float* wx2      = (const float*)d_in[9];
    const float* w_dw     = (const float*)d_in[10];
    const float* w_pw     = (const float*)d_in[11];
    const float* b_pw     = (const float*)d_in[12];
    float* out = (float*)d_out;

    char* ws = (char*)d_ws;
    int*      idx  = (int*)(ws);                              // 512 KB
    _Float16* fa   = (_Float16*)(ws +  1ull * 1048576);       // 50.3 MB
    _Float16* w0p  = (_Float16*)(ws + 52ull * 1048576);       // 1.5 MB
    _Float16* wpwp = (_Float16*)(ws + 54ull * 1048576);       // 0.4 MB
    float*    t_ws = (float*)   (ws + 55ull * 1048576);       // 8 MB
    _Float16* dwv  = (_Float16*)(ws + 64ull * 1048576);       // 12.6 MB

    pack_w0_kernel <<<3072, 256, 0, stream>>>(wx0, w0p);
    pack_wpw_kernel<<< 768, 256, 0, stream>>>(w_pw, wpwp);
    knn_kernel     <<<NPTS,  32, 0, stream>>>(center, coords, idx);
    build_fa_kernel<<<NPTS, 128, 0, stream>>>(coords, center, features,
                                              w_d1, b_d1, w_d2, b_d2, idx, fa);
    gemm_async_kernel<CINK, KT1 / 2, false><<<NPTS / 64, 256, 0, stream>>>(fa, w0p, nullptr, t_ws);
    middle_kernel  <<<NPTS, 256, 0, stream>>>(t_ws, fa, wx1, wx2, w_dw, dwv);
    gemm_async_kernel<CDW, KT2 / 2, true><<<NPTS / 64, 256, 0, stream>>>(dwv, wpwp, b_pw, out);
}
